// Attention_83356725281353
// MI455X (gfx1250) — matrix operand order repaired and checked by reading, not compiled, not run on hardware
//
#include <hip/hip_runtime.h>
#include <hip/hip_bf16.h>
#include <stdint.h>

#define B_   32
#define QN   4
#define DIM  2048
#define NH   16
#define HD   128
#define KVN  8192
#define NSLICE 8
#define SLICE (KVN / NSLICE)   // 1024
#define CHUNK 32
#define NCHUNK (SLICE / CHUNK) // 32

// dynamic LDS layout (bytes)
#define SMEM_K0 0
#define SMEM_K1 (CHUNK * HD * 4)
#define SMEM_V0 (2 * CHUNK * HD * 4)
#define SMEM_V1 (3 * CHUNK * HD * 4)
#define SMEM_VT (4 * CHUNK * HD * 4)
#define SMEM_P  (SMEM_VT + HD * CHUNK * 2)
#define SMEM_BYTES (SMEM_P + 4 * 16 * CHUNK * 2)   // 77,824 B

typedef __bf16 bf16_t;
typedef __attribute__((ext_vector_type(16))) __bf16 v16bf;
typedef __attribute__((ext_vector_type(8)))  float  v8f;
typedef __attribute__((ext_vector_type(4)))  unsigned int v4u;
typedef __attribute__((ext_vector_type(8)))  int v8i;
typedef __attribute__((ext_vector_type(4)))  int v4i;

union Frag { v16bf v; __bf16 h[16]; uint4 u[2]; };

__device__ __forceinline__ v8f wmma_bf16(const Frag& a, const Frag& b, v8f c) {
  return __builtin_amdgcn_wmma_f32_16x16x32_bf16(false, a.v, false, b.v, (short)0, c,
                                                 false, false);
}
// A-fragment: per-lane chunks at +0 and +16 (elements), 16B each
__device__ __forceinline__ void load_a(Frag& f, const bf16_t* p) {
  f.u[0] = *(const uint4*)p;
  f.u[1] = *(const uint4*)(p + 16);
}
// B-fragment: per-lane chunks at +0 and +16 (elements), 16B each
__device__ __forceinline__ void load_b(Frag& f, const bf16_t* p) {
  f.u[0] = *(const uint4*)p;
  f.u[1] = *(const uint4*)(p + 16);
}

#if defined(__gfx1250__) && __has_builtin(__builtin_amdgcn_tensor_load_to_lds)
#define HAVE_TDM 1
#endif

#ifdef HAVE_TDM
// 2D TDM: tile_rows x HD f32 tile, row stride HD elems, global -> LDS at lds_off.
__device__ __forceinline__ void tdm_load_rows(uint32_t lds_off, const float* gsrc,
                                              int tile_rows) {
  uint64_t ga = (uint64_t)(uintptr_t)gsrc;
  v4u g0 = {};
  g0[0] = 1u;                                               // count=1 (valid user D#)
  g0[1] = lds_off;                                          // lds_addr
  g0[2] = (uint32_t)ga;                                     // global_addr[31:0]
  g0[3] = (uint32_t)((ga >> 32) & 0x1FFFFFFu) | (2u << 30); // addr[56:32] | type=2
  v8i g1 = {};
  g1[0] = (int)(2u << 16);                                  // data_size = 4 bytes
  g1[1] = (int)((uint32_t)HD << 16);                        // tensor_dim0[15:0] = 128
  g1[2] = (int)(((uint32_t)tile_rows & 0xFFFFu) << 16);     // tensor_dim1[15:0]
  g1[3] = (int)((uint32_t)HD << 16);                        // tile_dim0 = 128
  g1[4] = (int)((uint32_t)tile_rows & 0xFFFFu);             // tile_dim1
  g1[5] = HD;                                               // tensor_dim0_stride = 128
  v4i g2 = {};
  v4i g3 = {};
#if __clang_major__ >= 23
  v8i g4 = {};
  __builtin_amdgcn_tensor_load_to_lds(g0, g1, g2, g3, g4, 0);
#else
  __builtin_amdgcn_tensor_load_to_lds(g0, g1, g2, g3, 0);
#endif
}
#endif

// ---------------- elementwise cast / transpose-cast ----------------
__global__ void cast_kernel(bf16_t* __restrict__ dst, const float* __restrict__ src, int n) {
  int i = blockIdx.x * 256 + threadIdx.x;
  if (i < n) dst[i] = (bf16_t)src[i];
}

// dst[n][k] = (bf16) src[k][n]   (src is K x N row-major f32)
__global__ void transpose_cast(bf16_t* __restrict__ dst, const float* __restrict__ src,
                               int K, int N) {
  int i = blockIdx.x * 256 + threadIdx.x;
  if (i >= N * K) return;
  int n = i / K, k = i - n * K;
  dst[(size_t)n * K + k] = (bf16_t)src[(size_t)k * N + n];
}

// ---------------- QKV projection GEMM: [128 x 2048] @ [2048 x 2304] ----------------
__global__ __launch_bounds__(128) void qkv_gemm(
    const bf16_t* __restrict__ xb, const bf16_t* __restrict__ wt,
    const float* __restrict__ bq, const float* __restrict__ bk,
    const float* __restrict__ bv,
    bf16_t* __restrict__ qmat, float* __restrict__ knew, float* __restrict__ vnew) {
  const int lane = threadIdx.x & 31;
  const int wid  = threadIdx.x >> 5;
  const int l16 = lane & 15, sel = lane >> 4;
  const int NTN = (DIM + 2 * HD) / 64;              // 36 column groups
  const int tile = blockIdx.x * 4 + wid;            // 0..287
  const int mt = tile / NTN;
  const int n0 = (tile % NTN) * 64;
  const int row = mt * 16 + l16;

  const bf16_t* ap = xb + (size_t)row * DIM + sel * 8;
  const bf16_t* bp[4];
#pragma unroll
  for (int nt = 0; nt < 4; ++nt)
    bp[nt] = wt + (size_t)(n0 + nt * 16 + l16) * DIM + sel * 8;

  v8f acc[4];
#pragma unroll
  for (int i = 0; i < 4; ++i) acc[i] = {};

  // two-stage software pipeline, unrolled by 2 (no register copies)
  Frag fa0, fa1, fb0[4], fb1[4];
  load_a(fa0, ap);
#pragma unroll
  for (int nt = 0; nt < 4; ++nt) load_b(fb0[nt], bp[nt]);

  for (int k0 = 0; k0 < DIM; k0 += 64) {
    // stage-1 loads (k0+32): always in range since DIM % 64 == 0
    load_a(fa1, ap + k0 + 32);
#pragma unroll
    for (int nt = 0; nt < 4; ++nt) load_b(fb1[nt], bp[nt] + k0 + 32);
#pragma unroll
    for (int nt = 0; nt < 4; ++nt) acc[nt] = wmma_bf16(fa0, fb0[nt], acc[nt]);
    if (k0 + 64 < DIM) {   // stage-0 loads for next iteration
      load_a(fa0, ap + k0 + 64);
#pragma unroll
      for (int nt = 0; nt < 4; ++nt) load_b(fb0[nt], bp[nt] + k0 + 64);
    }
#pragma unroll
    for (int nt = 0; nt < 4; ++nt) acc[nt] = wmma_bf16(fa1, fb1[nt], acc[nt]);
  }
#pragma unroll
  for (int nt = 0; nt < 4; ++nt) {
    const int n = n0 + nt * 16 + l16;
#pragma unroll
    for (int r = 0; r < 8; ++r) {
      const int m = mt * 16 + r + sel * 8;
      const int b = m >> 2, q = m & 3;
      float v = acc[nt][r];
      if (n < DIM) {
        const int h = n >> 7, e = n & 127;
        v += bq[n];
        qmat[(((size_t)b * NH + h) * QN + q) * HD + e] = (bf16_t)v;
      } else if (n < DIM + HD) {
        const int e = n - DIM;
        knew[((size_t)b * QN + q) * HD + e] = v + bk[e];
      } else {
        const int e = n - DIM - HD;
        vnew[((size_t)b * QN + q) * HD + e] = v + bv[e];
      }
    }
  }
}

// ---------------- Flash-decode attention (split-KV partials) ----------------
__global__ __launch_bounds__(128) void attn_fa(
    const bf16_t* __restrict__ qmat,
    const float* __restrict__ cache_k, const float* __restrict__ cache_v,
    const float* __restrict__ knew, const float* __restrict__ vnew,
    const float* __restrict__ bias,
    float* __restrict__ o_part, float* __restrict__ m_part, float* __restrict__ l_part) {
  extern __shared__ __align__(16) char smem[];
  bf16_t* sVT = (bf16_t*)(smem + SMEM_VT);

  const int b = blockIdx.x;
  const int s = blockIdx.y;
  const int tid = threadIdx.x;
  const int wid = tid >> 5;
  const int lane = tid & 31;
  const int l16 = lane & 15;
  const int sel = lane >> 4;
  const int h0 = wid * 4;                    // this wave owns heads h0..h0+3
  const float scale = 0.08838834764831845f;  // 1/sqrt(128)
  bf16_t* sPw = (bf16_t*)(smem + SMEM_P) + wid * 16 * CHUNK;
#ifdef HAVE_TDM
  const uint32_t smem_base = (uint32_t)(uintptr_t)smem;   // runtime LDS byte offset
#endif

  // A fragments: 16 rows = (4 heads x 4 queries), K = head_dim 128 (4 frags of 32)
  Frag fq[4];
  {
    const int h = h0 + (l16 >> 2);
    const int q = l16 & 3;
    const bf16_t* qp = qmat + (((size_t)b * NH + h) * QN + q) * HD;
#pragma unroll
    for (int e0c = 0; e0c < 4; ++e0c) load_a(fq[e0c], qp + e0c * 32 + sel * 8);
  }

  v8f oacc[8];
#pragma unroll
  for (int i = 0; i < 8; ++i) oacc[i] = {};
  float mrow[8], lrow[8];
#pragma unroll
  for (int r = 0; r < 8; ++r) { mrow[r] = -3.0e38f; lrow[r] = 0.f; }

#ifdef HAVE_TDM
  // prologue: kick off DMA of chunk 0 into buffer 0
  if (tid < 32) {
    const int j0 = s * SLICE;
    const int rc = min(CHUNK, (KVN - QN) - j0);
    tdm_load_rows(smem_base + SMEM_K0, cache_k + ((size_t)b * KVN + j0 + QN) * HD, rc);
    tdm_load_rows(smem_base + SMEM_V0, cache_v + ((size_t)b * KVN + j0 + QN) * HD, rc);
  }
#endif

  for (int c = 0; c < NCHUNK; ++c) {
    const int cur = c & 1;
    const int j0 = s * SLICE + c * CHUNK;
    const int rows_cache = min(CHUNK, (KVN - QN) - j0);
    float* sK = (float*)(smem + (cur ? SMEM_K1 : SMEM_K0));
    float* sV = (float*)(smem + (cur ? SMEM_V1 : SMEM_V0));
#ifdef HAVE_TDM
    __builtin_amdgcn_s_wait_tensorcnt(0);   // wave0: chunk c has landed
    __syncthreads();                        // chunk c visible; prev reads of buf[1-cur] done
    if (c + 1 < NCHUNK && tid < 32) {       // overlap DMA of chunk c+1 with compute of c
      const int j1 = j0 + CHUNK;
      const int rc1 = min(CHUNK, (KVN - QN) - j1);
      const uint32_t koff = smem_base + (cur ? SMEM_K0 : SMEM_K1);
      const uint32_t voff = smem_base + (cur ? SMEM_V0 : SMEM_V1);
      tdm_load_rows(koff, cache_k + ((size_t)b * KVN + j1 + QN) * HD, rc1);
      tdm_load_rows(voff, cache_v + ((size_t)b * KVN + j1 + QN) * HD, rc1);
    }
#else
    __syncthreads();
    {
      const float* gk = cache_k + ((size_t)b * KVN + j0 + QN) * HD;
      const float* gv = cache_v + ((size_t)b * KVN + j0 + QN) * HD;
      for (int idx = tid * 4; idx < rows_cache * HD; idx += 128 * 4) {
        *(float4*)&sK[idx] = *(const float4*)&gk[idx];
        *(float4*)&sV[idx] = *(const float4*)&gv[idx];
      }
    }
    __syncthreads();
#endif
    if (rows_cache < CHUNK) {  // rolled-in new tokens at the tail (last chunk only)
      for (int idx = tid; idx < (CHUNK - rows_cache) * HD; idx += 128) {
        int r = rows_cache + (idx >> 7);
        int e = idx & 127;
        int jj = j0 + r - (KVN - QN);
        sK[r * HD + e] = knew[((size_t)b * QN + jj) * HD + e];
        sV[r * HD + e] = vnew[((size_t)b * QN + jj) * HD + e];
      }
      __syncthreads();
    }
    // block-wide transpose+convert of V: sVT[e][kk] bf16 (conflict-free column reads)
    {
      const int e = tid;  // 0..127
#pragma unroll 4
      for (int kk = 0; kk < CHUNK; ++kk)
        sVT[e * CHUNK + kk] = (bf16_t)sV[kk * HD + e];
    }
    // prefetch next chunk's bias rows (one 128B line per row of this wave's tile)
    if (lane < 16 && c + 1 < NCHUNK) {
      const int hh = h0 + (lane >> 2);
      const int qq = lane & 3;
      __builtin_prefetch(&bias[(((size_t)b * NH + hh) * QN + qq) * KVN + j0 + CHUNK], 0, 1);
    }
    __syncthreads();

    // ---- scores: 16 x 32 tile via 8 WMMAs (K streamed f32->bf16 from LDS) ----
    v8f sc[2];
    float pmax[8];
#pragma unroll
    for (int r = 0; r < 8; ++r) pmax[r] = -3.0e38f;
#pragma unroll
    for (int ct = 0; ct < 2; ++ct) {
      v8f acc = {};
      const float* kp = &sK[(ct * 16 + l16) * HD + sel * 8];
#pragma unroll
      for (int e0c = 0; e0c < 4; ++e0c) {
        Frag fb;
#pragma unroll
        for (int t = 0; t < 16; ++t) fb.h[t] = (bf16_t)kp[e0c * 32 + t + (t & 8)];
        acc = wmma_bf16(fq[e0c], fb, acc);
      }
#pragma unroll
      for (int r = 0; r < 8; ++r) {
        const int m = r + sel * 8;
        const int hh = h0 + (m >> 2);
        const int qq = m & 3;
        const int j = j0 + ct * 16 + l16;
        float bvl = __builtin_nontemporal_load(
            &bias[(((size_t)b * NH + hh) * QN + qq) * KVN + j]);
        float v = acc[r] * scale + bvl;
        sc[ct][r] = v;
        pmax[r] = fmaxf(pmax[r], v);
      }
    }
    // ---- online softmax (row reductions across the 16-lane group) ----
#pragma unroll
    for (int r = 0; r < 8; ++r)
#pragma unroll
      for (int msk = 1; msk < 16; msk <<= 1)
        pmax[r] = fmaxf(pmax[r], __shfl_xor(pmax[r], msk, 32));
    float mscale[8], psum[8];
#pragma unroll
    for (int r = 0; r < 8; ++r) {
      float mn = fmaxf(mrow[r], pmax[r]);
      mscale[r] = __expf(mrow[r] - mn);
      mrow[r] = mn;
      psum[r] = 0.f;
    }
#pragma unroll
    for (int ct = 0; ct < 2; ++ct)
#pragma unroll
      for (int r = 0; r < 8; ++r) {
        float p = __expf(sc[ct][r] - mrow[r]);
        sc[ct][r] = p;
        psum[r] += p;
      }
#pragma unroll
    for (int r = 0; r < 8; ++r) {
#pragma unroll
      for (int msk = 1; msk < 16; msk <<= 1)
        psum[r] += __shfl_xor(psum[r], msk, 32);
      lrow[r] = lrow[r] * mscale[r] + psum[r];
    }
#pragma unroll
    for (int et = 0; et < 8; ++et)
#pragma unroll
      for (int r = 0; r < 8; ++r) oacc[et][r] *= mscale[r];

    // ---- probs C-layout -> A-layout via wave-private LDS ----
#pragma unroll
    for (int ct = 0; ct < 2; ++ct)
#pragma unroll
      for (int r = 0; r < 8; ++r)
        sPw[(r + sel * 8) * CHUNK + ct * 16 + l16] = (bf16_t)sc[ct][r];

    Frag fp;
    load_a(fp, &sPw[l16 * CHUNK + sel * 8]);
    // ---- O += P @ V : 8 WMMAs over the 8 e-tiles ----
#pragma unroll
    for (int et = 0; et < 8; ++et) {
      Frag fv;
      load_b(fv, &sVT[(et * 16 + l16) * CHUNK + sel * 8]);
      oacc[et] = wmma_bf16(fp, fv, oacc[et]);
    }
  }

  // ---- write slice partials ----
#pragma unroll
  for (int r = 0; r < 8; ++r) {
    const int m = r + sel * 8;
    const int hh = h0 + (m >> 2);
    const int qq = m & 3;
    const size_t base = (((size_t)s * B_ + b) * NH + hh) * QN + qq;
#pragma unroll
    for (int et = 0; et < 8; ++et)
      o_part[base * HD + et * 16 + l16] = oacc[et][r];
    if (l16 == 0) { m_part[base] = mrow[r]; l_part[base] = lrow[r]; }
  }
}

// ---------------- merge split-KV partials (log-sum-exp) ----------------
__global__ void merge_kernel(const float* __restrict__ o_part,
                             const float* __restrict__ m_part,
                             const float* __restrict__ l_part,
                             bf16_t* __restrict__ obf) {
  int i = blockIdx.x * 256 + threadIdx.x;
  if (i >= B_ * NH * QN * HD) return;
  int e = i & (HD - 1);
  int bhq = i >> 7;  // (b*16 + h)*4 + q
  float mg = -3.0e38f;
#pragma unroll
  for (int s = 0; s < NSLICE; ++s)
    mg = fmaxf(mg, m_part[(size_t)s * (B_ * NH * QN) + bhq]);
  float lg = 0.f, ov = 0.f;
#pragma unroll
  for (int s = 0; s < NSLICE; ++s) {
    size_t idx = (size_t)s * (B_ * NH * QN) + bhq;
    float w = __expf(m_part[idx] - mg);
    lg += w * l_part[idx];
    ov += w * o_part[idx * HD + e];
  }
  ov /= lg;
  int b = bhq >> 6, h = (bhq >> 2) & 15, q = bhq & 3;
  obf[((size_t)(b * QN + q)) * DIM + h * HD + e] = (bf16_t)ov;
}

// ---------------- output projection GEMM: [128 x 2048] @ [2048 x 2048] ----------------
__global__ __launch_bounds__(128) void out_gemm(
    const bf16_t* __restrict__ ob, const bf16_t* __restrict__ wot,
    const float* __restrict__ bo, float* __restrict__ out) {
  const int lane = threadIdx.x & 31;
  const int wid  = threadIdx.x >> 5;
  const int l16 = lane & 15, sel = lane >> 4;
  const int NTN = DIM / 64;                  // 32
  const int tile = blockIdx.x * 4 + wid;     // 0..255
  const int mt = tile / NTN;
  const int n0 = (tile % NTN) * 64;
  const int row = mt * 16 + l16;

  const bf16_t* ap = ob + (size_t)row * DIM + sel * 8;
  const bf16_t* bp[4];
#pragma unroll
  for (int nt = 0; nt < 4; ++nt)
    bp[nt] = wot + (size_t)(n0 + nt * 16 + l16) * DIM + sel * 8;

  v8f acc[4];
#pragma unroll
  for (int i = 0; i < 4; ++i) acc[i] = {};

  Frag fa0, fa1, fb0[4], fb1[4];
  load_a(fa0, ap);
#pragma unroll
  for (int nt = 0; nt < 4; ++nt) load_b(fb0[nt], bp[nt]);

  for (int k0 = 0; k0 < DIM; k0 += 64) {
    load_a(fa1, ap + k0 + 32);
#pragma unroll
    for (int nt = 0; nt < 4; ++nt) load_b(fb1[nt], bp[nt] + k0 + 32);
#pragma unroll
    for (int nt = 0; nt < 4; ++nt) acc[nt] = wmma_bf16(fa0, fb0[nt], acc[nt]);
    if (k0 + 64 < DIM) {
      load_a(fa0, ap + k0 + 64);
#pragma unroll
      for (int nt = 0; nt < 4; ++nt) load_b(fb0[nt], bp[nt] + k0 + 64);
    }
#pragma unroll
    for (int nt = 0; nt < 4; ++nt) acc[nt] = wmma_bf16(fa1, fb1[nt], acc[nt]);
  }
#pragma unroll
  for (int nt = 0; nt < 4; ++nt) {
    const int n = n0 + nt * 16 + l16;
#pragma unroll
    for (int r = 0; r < 8; ++r) {
      const int m = mt * 16 + r + sel * 8;
      out[(size_t)m * DIM + n] = acc[nt][r] + bo[n];
    }
  }
}

extern "C" void kernel_launch(void* const* d_in, const int* in_sizes, int n_in,
                              void* d_out, int out_size, void* d_ws, size_t ws_size,
                              hipStream_t stream) {
  (void)in_sizes; (void)n_in; (void)out_size; (void)ws_size;
  const float* x    = (const float*)d_in[0];
  const float* bias = (const float*)d_in[1];
  const float* ck   = (const float*)d_in[2];
  const float* cv   = (const float*)d_in[3];
  const float* wq   = (const float*)d_in[4];
  const float* bq   = (const float*)d_in[5];
  const float* wk   = (const float*)d_in[6];
  const float* bk   = (const float*)d_in[7];
  const float* wv   = (const float*)d_in[8];
  const float* bv   = (const float*)d_in[9];
  const float* wo   = (const float*)d_in[10];
  const float* bo   = (const float*)d_in[11];
  float* out = (float*)d_out;

  char* ws = (char*)d_ws;
  size_t off = 0;
  auto take = [&](size_t bytes) {
    char* p = ws + off;
    off = (off + bytes + 255) & ~(size_t)255;
    return p;
  };
  bf16_t* xb     = (bf16_t*)take((size_t)B_ * QN * DIM * 2);
  bf16_t* wqkvT  = (bf16_t*)take((size_t)(DIM + 2 * HD) * DIM * 2);
  bf16_t* woT    = (bf16_t*)take((size_t)DIM * DIM * 2);
  bf16_t* qmat   = (bf16_t*)take((size_t)B_ * NH * QN * HD * 2);
  float*  knew   = (float*) take((size_t)B_ * QN * HD * 4);
  float*  vnew   = (float*) take((size_t)B_ * QN * HD * 4);
  float*  o_part = (float*) take((size_t)NSLICE * B_ * NH * QN * HD * 4);
  float*  m_part = (float*) take((size_t)NSLICE * B_ * NH * QN * 4);
  float*  l_part = (float*) take((size_t)NSLICE * B_ * NH * QN * 4);
  bf16_t* obf    = (bf16_t*)take((size_t)B_ * QN * DIM * 2);

  cast_kernel<<<(B_ * QN * DIM) / 256, 256, 0, stream>>>(xb, x, B_ * QN * DIM);
  transpose_cast<<<(DIM * DIM) / 256, 256, 0, stream>>>(wqkvT, wq, DIM, DIM);
  transpose_cast<<<(HD * DIM) / 256, 256, 0, stream>>>(wqkvT + (size_t)DIM * DIM, wk, DIM, HD);
  transpose_cast<<<(HD * DIM) / 256, 256, 0, stream>>>(wqkvT + (size_t)(DIM + HD) * DIM, wv, DIM, HD);
  transpose_cast<<<(DIM * DIM) / 256, 256, 0, stream>>>(woT, wo, DIM, DIM);

  qkv_gemm<<<72, 128, 0, stream>>>(xb, wqkvT, bq, bk, bv, qmat, knew, vnew);
  attn_fa<<<dim3(B_, NSLICE), 128, SMEM_BYTES, stream>>>(qmat, ck, cv, knew, vnew, bias,
                                                         o_part, m_part, l_part);
  merge_kernel<<<(B_ * NH * QN * HD) / 256, 256, 0, stream>>>(o_part, m_part, l_part, obf);
  out_gemm<<<64, 128, 0, stream>>>(obf, woT, bo, out);
}
